// AttentionBlock_9766755631834
// MI455X (gfx1250) — matrix operand order repaired and checked by reading, not compiled, not run on hardware
//
#include <hip/hip_runtime.h>
#include <hip/hip_bf16.h>

// ---------------------------------------------------------------------------
// CDNA5 (gfx1250, wave32) attention block:
//   GroupNorm(32) -> 1x1 QKV -> 8-head softmax attention (flash-style)
//   -> 1x1 proj -> +residual
// Shapes: b=8, c=512, h=w=32 (n=1024), HEADS=8, d=64, scale=0.125
// All matrix math via v_wmma_f32_16x16x32_f16 (fp32 accumulate).
// GEMMs register-tiled 32x32 per wave: 4 WMMAs per K-step share 2 A + 2 B
// fragments -> halves L2 fragment traffic vs one-tile-per-wave.
// ---------------------------------------------------------------------------

typedef __attribute__((ext_vector_type(16))) _Float16 v16h;
typedef __attribute__((ext_vector_type(8)))  float    v8f;

#define NB   8
#define NC   512
#define NN   1024
#define NH   8
#define ND   64
#define NG   32

__device__ __forceinline__ v8f wmma_f16(v16h a, v16h b, v8f c) {
  // (neg_a, A, neg_b, B, c_mod, C, reuse_a, reuse_b)
  return __builtin_amdgcn_wmma_f32_16x16x32_f16(false, a, false, b, (short)0, c,
                                                false, false);
}

// A-matrix 16x32 f16 fragment from row-major [rows][ld] memory.
// ISA layout: lanes 0-15 hold M=lane, K={0..7,16..23}; lanes 16-31 K={8..15,24..31}.
__device__ __forceinline__ v16h load_a_frag(const _Float16* __restrict__ base,
                                            int row0, int k0, int ld) {
  int lane = threadIdx.x & 31;
  int m = lane & 15;
  int koff = (lane < 16) ? 0 : 8;
  const _Float16* p = base + (size_t)(row0 + m) * ld + k0 + koff;
  union { v16h v; uint4 u[2]; } f;
  f.u[0] = *(const uint4*)(p);
  f.u[1] = *(const uint4*)(p + 16);
  return f.v;
}

// B-matrix 32x16 f16 fragment: B[kk][nn] with storage transposed so that
// column nn is a contiguous row: baseT[(n0+nn)*ld + k].
// ISA layout: lane = nn (mod 16); lanes 0-15 hold K={0..7,16..23}, lanes 16-31 K={8..15,24..31}.
__device__ __forceinline__ v16h load_b_frag(const _Float16* __restrict__ baseT,
                                            int n0, int k0, int ld) {
  int lane = threadIdx.x & 31;
  const _Float16* p = baseT + (size_t)(n0 + (lane & 15)) * ld + k0 +
                      ((lane < 16) ? 0 : 8);
  union { v16h v; uint4 u[2]; } f;
  f.u[0] = *(const uint4*)(p);
  f.u[1] = *(const uint4*)(p + 16);
  return f.v;
}

// ---------------------------------------------------------------------------
// Kernel 0: pack fp32 weights -> f16 (row-major kept)
// ---------------------------------------------------------------------------
__global__ void pack_w_kernel(const float* __restrict__ qkv_w,
                              const float* __restrict__ proj_w,
                              _Float16* __restrict__ qw16,
                              _Float16* __restrict__ pw16) {
  int i = blockIdx.x * blockDim.x + threadIdx.x;
  if (i < 3 * NC * NC) qw16[i] = (_Float16)qkv_w[i];
  if (i < NC * NC)     pw16[i] = (_Float16)proj_w[i];
}

// ---------------------------------------------------------------------------
// Kernel 1: GroupNorm + affine, write y transposed [b][n][c] in f16
// one block per (b, group): 16 channels x 1024 spatial = 16384 elements
// ---------------------------------------------------------------------------
__global__ __launch_bounds__(256) void gn_kernel(const float* __restrict__ x,
                                                 const float* __restrict__ nw,
                                                 const float* __restrict__ nb,
                                                 _Float16* __restrict__ yT) {
  int b = blockIdx.x >> 5;
  int g = blockIdx.x & 31;
  const float* xp = x + (size_t)(b * NC + g * 16) * NN;

  __shared__ float s_sum[256], s_sq[256];
  __shared__ float sw[16], sb[16];
  if (threadIdx.x < 16) {
    sw[threadIdx.x] = nw[g * 16 + threadIdx.x];
    sb[threadIdx.x] = nb[g * 16 + threadIdx.x];
  }
  float sum = 0.f, sq = 0.f;
  for (int i = threadIdx.x; i < 16 * NN; i += 256) {
    float v = xp[i];
    sum += v; sq += v * v;
  }
  s_sum[threadIdx.x] = sum; s_sq[threadIdx.x] = sq;
  __syncthreads();
  for (int s = 128; s > 0; s >>= 1) {
    if (threadIdx.x < s) {
      s_sum[threadIdx.x] += s_sum[threadIdx.x + s];
      s_sq[threadIdx.x]  += s_sq[threadIdx.x + s];
    }
    __syncthreads();
  }
  float mean = s_sum[0] * (1.f / 16384.f);
  float var  = s_sq[0] * (1.f / 16384.f) - mean * mean;
  float inv  = rsqrtf(var + 1e-5f);

  _Float16* yp = yT + (size_t)b * NN * NC + g * 16;
  for (int i = threadIdx.x; i < 16 * NN; i += 256) {
    int ch = i >> 10, n = i & (NN - 1);
    float v = (xp[i] - mean) * inv * sw[ch] + sb[ch];
    yp[(size_t)n * NC + ch] = (_Float16)v;
  }
}

// ---------------------------------------------------------------------------
// QKV epilogue helper: scatter one 16x16 C tile into Qt/Kt/V layouts.
//   Qt [b][h][n][d] (pre-scaled by 0.125), Kt [b][h][n][d], V [b][h][d][n]
// ---------------------------------------------------------------------------
__device__ __forceinline__ void qkv_store_tile(v8f acc, int b, int obase,
                                               int nbase, int lane,
                                               const float* __restrict__ bias,
                                               _Float16* __restrict__ Qt,
                                               _Float16* __restrict__ Kt,
                                               _Float16* __restrict__ V) {
  int o0  = obase + ((lane < 16) ? 0 : 8);   // this lane's 8 rows: o0..o0+7
  int col = nbase + (lane & 15);
  float vals[8];
#pragma unroll
  for (int r = 0; r < 8; ++r) vals[r] = acc[r] + bias[o0 + r];

  if (o0 < NC) {                     // Q: scale + packed 16B store
    int head = o0 >> 6, dd0 = o0 & 63;
    union { uint4 u; _Float16 h[8]; } pk;
#pragma unroll
    for (int r = 0; r < 8; ++r) pk.h[r] = (_Float16)(vals[r] * 0.125f);
    *(uint4*)(Qt + (((size_t)(b * NH + head) * NN + col) * ND + dd0)) = pk.u;
  } else if (o0 < 2 * NC) {          // K: packed 16B store
    int oo = o0 - NC;
    int head = oo >> 6, dd0 = oo & 63;
    union { uint4 u; _Float16 h[8]; } pk;
#pragma unroll
    for (int r = 0; r < 8; ++r) pk.h[r] = (_Float16)vals[r];
    *(uint4*)(Kt + (((size_t)(b * NH + head) * NN + col) * ND + dd0)) = pk.u;
  } else {                           // V: [d][n] scattered f16 stores
    int oo = o0 - 2 * NC;
    int head = oo >> 6, dd0 = oo & 63;
#pragma unroll
    for (int r = 0; r < 8; ++r)
      V[((size_t)(b * NH + head) * ND + dd0 + r) * NN + col] = (_Float16)vals[r];
  }
}

// ---------------------------------------------------------------------------
// Kernel 2: QKV GEMM  qkv[o,n] = qkv_w[o,:] . y[:,n] + bias
// One wave -> 32x32 output tile (2x2 WMMA tiles); M=1536, N=1024, K=512.
// Per K-step: 2 A-frags + 2 B-frags feed 4 WMMAs (1 KB L2 traffic per WMMA).
// ---------------------------------------------------------------------------
__global__ __launch_bounds__(256) void qkv_gemm_kernel(
    const _Float16* __restrict__ w16, const _Float16* __restrict__ yT,
    const float* __restrict__ bias, _Float16* __restrict__ Qt,
    _Float16* __restrict__ Kt, _Float16* __restrict__ V) {
  int wid  = blockIdx.x * 8 + (threadIdx.x >> 5);
  int lane = threadIdx.x & 31;
  int b    = wid / (48 * 32);
  int rem  = wid % (48 * 32);
  int obase = (rem / 32) * 32;
  int nbase = (rem % 32) * 32;
  const _Float16* Bb = yT + (size_t)b * NN * NC;

  v8f acc00 = {}, acc01 = {}, acc10 = {}, acc11 = {};
#pragma unroll
  for (int k0 = 0; k0 < NC; k0 += 32) {
    v16h a0 = load_a_frag(w16, obase,      k0, NC);
    v16h a1 = load_a_frag(w16, obase + 16, k0, NC);
    v16h b0 = load_b_frag(Bb,  nbase,      k0, NC);
    v16h b1 = load_b_frag(Bb,  nbase + 16, k0, NC);
    acc00 = wmma_f16(a0, b0, acc00);
    acc01 = wmma_f16(a0, b1, acc01);
    acc10 = wmma_f16(a1, b0, acc10);
    acc11 = wmma_f16(a1, b1, acc11);
  }
  qkv_store_tile(acc00, b, obase,      nbase,      lane, bias, Qt, Kt, V);
  qkv_store_tile(acc01, b, obase,      nbase + 16, lane, bias, Qt, Kt, V);
  qkv_store_tile(acc10, b, obase + 16, nbase,      lane, bias, Qt, Kt, V);
  qkv_store_tile(acc11, b, obase + 16, nbase + 16, lane, bias, Qt, Kt, V);
}

// ---------------------------------------------------------------------------
// Kernel 3: flash attention per (b,head). Wave owns 16 query rows; loops over
// keys in steps of 32 with online softmax; P goes through per-wave LDS to
// convert C-layout -> A-layout. Output written transposed At [b][n][c] f16.
// ---------------------------------------------------------------------------
__global__ __launch_bounds__(256) void attn_kernel(
    const _Float16* __restrict__ Qt, const _Float16* __restrict__ Kt,
    const _Float16* __restrict__ V, _Float16* __restrict__ At) {
  __shared__ __align__(16) _Float16 pbuf[8][16 * 32];
  int wv   = threadIdx.x >> 5;
  int lane = threadIdx.x & 31;
  int bh   = blockIdx.x >> 3;          // 0..63 = b*8+head
  int qblk = blockIdx.x & 7;
  int b = bh >> 3, head = bh & 7;
  int qbase = qblk * 128 + wv * 16;

  const _Float16* Qbh = Qt + (size_t)bh * NN * ND;
  const _Float16* Kbh = Kt + (size_t)bh * NN * ND;
  const _Float16* Vbh = V  + (size_t)bh * ND * NN;

  v16h Aq0 = load_a_frag(Qbh, qbase, 0, ND);
  v16h Aq1 = load_a_frag(Qbh, qbase, 32, ND);

  float mst[8], lst[8];
  v8f O0 = {}, O1 = {}, O2 = {}, O3 = {};
#pragma unroll
  for (int r = 0; r < 8; ++r) { mst[r] = -3.0e38f; lst[r] = 0.f; }

  _Float16* pb = &pbuf[wv][0];
  int c0 = lane & 15;
  int moff = (lane < 16) ? 0 : 8;

  for (int j0 = 0; j0 < NN; j0 += 32) {
    // keep the K/V stream ahead of the 8-WMMA body (global_prefetch_b8)
    if (j0 + 32 < NN) {
      __builtin_prefetch(Kbh + (size_t)(j0 + 32 + (lane & 15)) * ND, 0, 3);
      __builtin_prefetch(Vbh + (size_t)(lane & 15) * NN + j0 + 32, 0, 3);
    }

    v8f S0 = {}, S1 = {};
    S0 = wmma_f16(Aq0, load_b_frag(Kbh, j0, 0, ND), S0);
    S0 = wmma_f16(Aq1, load_b_frag(Kbh, j0, 32, ND), S0);
    S1 = wmma_f16(Aq0, load_b_frag(Kbh, j0 + 16, 0, ND), S1);
    S1 = wmma_f16(Aq1, load_b_frag(Kbh, j0 + 16, 32, ND), S1);

#pragma unroll
    for (int r = 0; r < 8; ++r) {
      float s0 = S0[r], s1 = S1[r];
      float mx = fmaxf(s0, s1);
      mx = fmaxf(mx, __shfl_xor(mx, 1, 32));
      mx = fmaxf(mx, __shfl_xor(mx, 2, 32));
      mx = fmaxf(mx, __shfl_xor(mx, 4, 32));
      mx = fmaxf(mx, __shfl_xor(mx, 8, 32));
      float mnew = fmaxf(mst[r], mx);
      float corr = __expf(mst[r] - mnew);
      float p0 = __expf(s0 - mnew);
      float p1 = __expf(s1 - mnew);
      float ps = p0 + p1;
      ps += __shfl_xor(ps, 1, 32);
      ps += __shfl_xor(ps, 2, 32);
      ps += __shfl_xor(ps, 4, 32);
      ps += __shfl_xor(ps, 8, 32);
      lst[r] = lst[r] * corr + ps;
      mst[r] = mnew;
      O0[r] *= corr; O1[r] *= corr; O2[r] *= corr; O3[r] *= corr;
      int m = r + moff;
      pb[m * 32 + c0]      = (_Float16)p0;   // C-layout -> row-major P tile
      pb[m * 32 + 16 + c0] = (_Float16)p1;
    }

    // P (16x32 f16) back out of LDS in A-fragment layout (per-wave, in-order)
    v16h Ap;
    {
      union { v16h v; uint4 u[2]; } f;
      const _Float16* pp = pb + (lane & 15) * 32 + ((lane < 16) ? 0 : 8);
      f.u[0] = *(const uint4*)(pp);
      f.u[1] = *(const uint4*)(pp + 16);
      Ap = f.v;
    }
    O0 = wmma_f16(Ap, load_b_frag(Vbh, 0, j0, NN), O0);
    O1 = wmma_f16(Ap, load_b_frag(Vbh, 16, j0, NN), O1);
    O2 = wmma_f16(Ap, load_b_frag(Vbh, 32, j0, NN), O2);
    O3 = wmma_f16(Ap, load_b_frag(Vbh, 48, j0, NN), O3);
  }

  _Float16* Ab = At + (size_t)b * NN * NC + head * ND;
#pragma unroll
  for (int r = 0; r < 8; ++r) {
    float inv = 1.0f / lst[r];
    int n = qbase + r + moff;
    _Float16* row = Ab + (size_t)n * NC;
    row[c0]      = (_Float16)(O0[r] * inv);
    row[16 + c0] = (_Float16)(O1[r] * inv);
    row[32 + c0] = (_Float16)(O2[r] * inv);
    row[48 + c0] = (_Float16)(O3[r] * inv);
  }
}

// ---------------------------------------------------------------------------
// Kernel 4: proj GEMM + bias + residual, fp32 out [b][c][n]
// One wave -> 32x32 output tile (2x2 WMMA tiles); M=512, N=1024, K=512.
// ---------------------------------------------------------------------------
__global__ __launch_bounds__(256) void proj_gemm_kernel(
    const _Float16* __restrict__ pw16, const _Float16* __restrict__ At,
    const float* __restrict__ bias, const float* __restrict__ x,
    float* __restrict__ out) {
  int wid  = blockIdx.x * 8 + (threadIdx.x >> 5);
  int lane = threadIdx.x & 31;
  int b    = wid / (16 * 32);
  int rem  = wid % (16 * 32);
  int obase = (rem / 32) * 32;
  int nbase = (rem % 32) * 32;
  const _Float16* Bb = At + (size_t)b * NN * NC;

  v8f acc00 = {}, acc01 = {}, acc10 = {}, acc11 = {};
#pragma unroll
  for (int k0 = 0; k0 < NC; k0 += 32) {
    v16h a0 = load_a_frag(pw16, obase,      k0, NC);
    v16h a1 = load_a_frag(pw16, obase + 16, k0, NC);
    v16h b0 = load_b_frag(Bb,   nbase,      k0, NC);
    v16h b1 = load_b_frag(Bb,   nbase + 16, k0, NC);
    acc00 = wmma_f16(a0, b0, acc00);
    acc01 = wmma_f16(a0, b1, acc01);
    acc10 = wmma_f16(a1, b0, acc10);
    acc11 = wmma_f16(a1, b1, acc11);
  }

  v8f accs[4] = {acc00, acc01, acc10, acc11};
#pragma unroll
  for (int t = 0; t < 4; ++t) {
    int o0  = obase + ((t >> 1) ? 16 : 0) + ((lane < 16) ? 0 : 8);
    int col = nbase + ((t & 1) ? 16 : 0) + (lane & 15);
#pragma unroll
    for (int r = 0; r < 8; ++r) {
      size_t idx = ((size_t)b * NC + o0 + r) * NN + col;
      out[idx] = accs[t][r] + bias[o0 + r] + x[idx];
    }
  }
}

// ---------------------------------------------------------------------------
extern "C" void kernel_launch(void* const* d_in, const int* in_sizes, int n_in,
                              void* d_out, int out_size, void* d_ws, size_t ws_size,
                              hipStream_t stream) {
  const float* x      = (const float*)d_in[0];
  const float* norm_w = (const float*)d_in[1];
  const float* norm_b = (const float*)d_in[2];
  const float* qkv_w  = (const float*)d_in[3];
  const float* qkv_b  = (const float*)d_in[4];
  const float* proj_w = (const float*)d_in[5];
  const float* proj_b = (const float*)d_in[6];
  float* out = (float*)d_out;

  // workspace carve-up (all f16, all 256B-aligned offsets); total ~42 MB,
  // entirely resident in MI455X's 192 MB L2 across kernels.
  char* ws = (char*)d_ws;
  const size_t SZ_Y  = (size_t)NB * NN * NC * 2;        // y^T   [b][n][c]
  const size_t SZ_QW = (size_t)3 * NC * NC * 2;         // qkv_w f16
  const size_t SZ_PW = (size_t)NC * NC * 2;             // proj_w f16
  const size_t SZ_T  = (size_t)NB * NH * NN * ND * 2;   // Qt / Kt / V / At each
  _Float16* yT   = (_Float16*)(ws);
  _Float16* qw16 = (_Float16*)(ws + SZ_Y);
  _Float16* pw16 = (_Float16*)(ws + SZ_Y + SZ_QW);
  _Float16* Qt   = (_Float16*)(ws + SZ_Y + SZ_QW + SZ_PW);
  _Float16* Kt   = (_Float16*)(ws + SZ_Y + SZ_QW + SZ_PW + SZ_T);
  _Float16* Vd   = (_Float16*)(ws + SZ_Y + SZ_QW + SZ_PW + 2 * SZ_T);
  _Float16* At   = (_Float16*)(ws + SZ_Y + SZ_QW + SZ_PW + 3 * SZ_T);

  // 0) pack weights to f16
  pack_w_kernel<<<(3 * NC * NC + 255) / 256, 256, 0, stream>>>(qkv_w, proj_w,
                                                               qw16, pw16);
  // 1) GroupNorm -> y^T f16
  gn_kernel<<<NB * NG, 256, 0, stream>>>(x, norm_w, norm_b, yT);
  // 2) QKV GEMM: 8 batches x (48 x 32) 32x32 wave tiles, 8 waves/block
  qkv_gemm_kernel<<<NB * 48 * 32 / 8, 256, 0, stream>>>(qw16, yT, qkv_b,
                                                        Qt, Kt, Vd);
  // 3) flash attention: 64 (b,h) x 8 query-blocks of 128 rows
  attn_kernel<<<64 * 8, 256, 0, stream>>>(Qt, Kt, Vd, At);
  // 4) proj GEMM + residual: 8 batches x (16 x 32) 32x32 wave tiles
  proj_gemm_kernel<<<NB * 16 * 32 / 8, 256, 0, stream>>>(pw16, At, proj_b,
                                                         x, out);
}
